// FlashLinearAttention_24292335026333
// MI455X (gfx1250) — matrix operand order repaired and checked by reading, not compiled, not run on hardware
//
#include <hip/hip_runtime.h>
#include <hip/hip_bf16.h>

// ---- problem constants -----------------------------------------------------
#define BB   2
#define LL   8192
#define EE   1024
#define HH   16
#define DD   64
#define CC   256
#define NCH  (LL / CC)         // 32 chunks
#define MM   (BB * LL)         // 16384 rows
#define NBH  (BB * HH)         // 32
#define EPSA 1e-6f
#define LNE  1e-5f

typedef __attribute__((ext_vector_type(16))) _Float16 v16h;
typedef __attribute__((ext_vector_type(8)))  _Float16 v8h;
typedef __attribute__((ext_vector_type(8)))  float    v8f;

// ---- WMMA helpers ----------------------------------------------------------
__device__ __forceinline__ v8f wmma32(v16h a, v16h b, v8f c) {
  return __builtin_amdgcn_wmma_f32_16x16x32_f16(false, a, false, b, (short)0, c,
                                                false, false);
}

// A fragment 16x32 from row-major f16 (ld halves per row).
// lane<16 -> K in {0..7, 16..23}; lane>=16 -> K in {8..15, 24..31}.
__device__ __forceinline__ v16h load_a_frag(const _Float16* base, int ld) {
  int lane = threadIdx.x & 31;
  int row  = lane & 15;
  int kb   = (lane >> 4) * 8;
  const _Float16* p = base + (size_t)row * ld;
  v8h lo = *(const v8h*)(p + kb);
  v8h hi = *(const v8h*)(p + 16 + kb);
  v16h a;
#pragma unroll
  for (int i = 0; i < 8; ++i) { a[i] = lo[i]; a[i + 8] = hi[i]; }
  return a;
}

// B fragment 32x16 from a TRANSPOSED [n][k] f16 buffer (ld halves per n-row).
__device__ __forceinline__ v16h load_b_fragT(const _Float16* baseT, int ld) {
  int lane = threadIdx.x & 31;
  int n    = lane & 15;
  int kb   = (lane >> 4) * 8;
  const _Float16* p = baseT + (size_t)n * ld + kb;
  v8h lo = *(const v8h*)(p);
  v8h hi = *(const v8h*)(p + 16);
  v16h b;
#pragma unroll
  for (int i = 0; i < 8; ++i) { b[i] = lo[i]; b[i + 8] = hi[i]; }
  return b;
}

// ---- CDNA5 async global->LDS copy (ASYNCcnt-tracked, bypasses VGPRs) -------
__device__ __forceinline__ void async_g2l_b128(uint32_t lds_off, uint64_t gaddr) {
  asm volatile("global_load_async_to_lds_b128 %0, %1, off"
               :: "v"(lds_off), "v"(gaddr) : "memory");
}
__device__ __forceinline__ void async_wait0() {
  asm volatile("s_wait_asynccnt 0" ::: "memory");
}

// branchless phi: elu(x)+1 = max(x,0) + exp(min(x,0))
__device__ __forceinline__ float phi_act(float x) {
  return fmaxf(x, 0.f) + __expf(fminf(x, 0.f));
}

// ---- kernel 1: fused QKV projection + phi, f32 in -> f16 out ---------------
// grid = (MM/128, EE/64, 3), block = 256
__global__ void la_proj_qkv(const float* __restrict__ x,
                            const float* __restrict__ Wq, const float* __restrict__ bq,
                            const float* __restrict__ Wk, const float* __restrict__ bk,
                            const float* __restrict__ Wv, const float* __restrict__ bv,
                            _Float16* __restrict__ qo, _Float16* __restrict__ ko,
                            _Float16* __restrict__ vo) {
  const float* W; const float* bias; _Float16* out; bool dophi;
  if (blockIdx.z == 0)      { W = Wq; bias = bq; out = qo; dophi = true;  }
  else if (blockIdx.z == 1) { W = Wk; bias = bk; out = ko; dophi = true;  }
  else                      { W = Wv; bias = bv; out = vo; dophi = false; }

  __shared__ _Float16 At[128 * 32];  // A tile row-major [m][k]
  __shared__ _Float16 Bt[64 * 32];   // B tile transposed [n][k]

  const int row0 = blockIdx.x * 128;
  const int col0 = blockIdx.y * 64;
  const int tid  = threadIdx.x;
  const int lane = tid & 31;
  const int wid  = tid >> 5;
  const int mt2  = wid & 3;          // covers M subtiles {2*mt2, 2*mt2+1}
  const int nh   = wid >> 2;         // covers N subtiles {2*nh, 2*nh+1}

  v8f cacc[2][2] = {};

  for (int kb = 0; kb < EE; kb += 32) {
    {  // stage A: 128x32 f32 -> f16, 16 values/thread
      int r = tid >> 1, cg = (tid & 1) * 16;
      const float4* px = (const float4*)(x + (size_t)(row0 + r) * EE + kb + cg);
      float4 f0 = px[0], f1 = px[1], f2 = px[2], f3 = px[3];
      _Float16* d = At + r * 32 + cg;
      d[0]  = (_Float16)f0.x; d[1]  = (_Float16)f0.y; d[2]  = (_Float16)f0.z; d[3]  = (_Float16)f0.w;
      d[4]  = (_Float16)f1.x; d[5]  = (_Float16)f1.y; d[6]  = (_Float16)f1.z; d[7]  = (_Float16)f1.w;
      d[8]  = (_Float16)f2.x; d[9]  = (_Float16)f2.y; d[10] = (_Float16)f2.z; d[11] = (_Float16)f2.w;
      d[12] = (_Float16)f3.x; d[13] = (_Float16)f3.y; d[14] = (_Float16)f3.z; d[15] = (_Float16)f3.w;
    }
    {  // stage B transposed: Bt[n][k] = W[kb+k][col0+n]
      int n = tid & 63, k0 = (tid >> 6) * 8;
#pragma unroll
      for (int i = 0; i < 8; ++i)
        Bt[n * 32 + k0 + i] = (_Float16)W[(size_t)(kb + k0 + i) * EE + col0 + n];
    }
    __syncthreads();
    v16h a0 = load_a_frag(At + (mt2 * 32) * 32, 32);
    v16h a1 = load_a_frag(At + (mt2 * 32 + 16) * 32, 32);
    v16h b0 = load_b_fragT(Bt + (nh * 32) * 32, 32);
    v16h b1 = load_b_fragT(Bt + (nh * 32 + 16) * 32, 32);
    cacc[0][0] = wmma32(a0, b0, cacc[0][0]);
    cacc[0][1] = wmma32(a0, b1, cacc[0][1]);
    cacc[1][0] = wmma32(a1, b0, cacc[1][0]);
    cacc[1][1] = wmma32(a1, b1, cacc[1][1]);
    __syncthreads();
  }

  const int cn  = lane & 15;
  const int rb2 = (lane >> 4) * 8;
#pragma unroll
  for (int mi = 0; mi < 2; ++mi) {
    int gr0 = row0 + (mt2 * 2 + mi) * 16 + rb2;
#pragma unroll
    for (int half = 0; half < 2; ++half) {
      const v8f cc = cacc[mi][half];
      int   gc  = col0 + nh * 32 + half * 16 + cn;
      float bv_ = bias[gc];
#pragma unroll
      for (int j = 0; j < 8; ++j) {
        float val = cc[j] + bv_;
        if (dophi) val = phi_act(val);
        out[(size_t)(gr0 + j) * EE + gc] = (_Float16)val;
      }
    }
  }
}

// ---- kernel 2a: per-chunk dS = k^T v (ST layout [e][d]) and dz = colsum(k) --
// grid = (NCH, NBH), block = 256, dynamic LDS 64KB
#define KV_LDS_BYTES 65536
__global__ void la_chunk_kv(const _Float16* __restrict__ k,
                            const _Float16* __restrict__ v,
                            float* __restrict__ dS, float* __restrict__ dz) {
  extern __shared__ char smem[];
  _Float16* kiT = (_Float16*)smem;             // [64][256]
  _Float16* viT = (_Float16*)(smem + 32768);   // [64][256]

  const int n  = blockIdx.x;
  const int bh = blockIdx.y;
  const int b  = bh >> 4, h = bh & 15;
  const size_t cb = (size_t)b * LL * EE + (size_t)h * DD + (size_t)n * CC * EE;
  const _Float16* kc = k + cb;
  const _Float16* vc = v + cb;

  const int tid  = threadIdx.x;
  const int lane = tid & 31;
  const int wid  = tid >> 5;
  const int cn   = lane & 15;
  const int rbv  = (lane >> 4) * 8;

  {  // transpose-stage k, v chunks
    const _Float16* kr = kc + (size_t)tid * EE;
    const _Float16* vr = vc + (size_t)tid * EE;
#pragma unroll 8
    for (int d = 0; d < DD; ++d) {
      kiT[d * CC + tid] = kr[d];
      viT[d * CC + tid] = vr[d];
    }
  }
  __syncthreads();

  const size_t base = ((size_t)n * NBH + bh) * (DD * DD);
  for (int t = wid; t < 16; t += 8) {
    int mt = t >> 2, et = t & 3;   // m = d subtile, n = e subtile
    v8f acc = {};
#pragma unroll
    for (int kcc = 0; kcc < CC; kcc += 32)
      acc = wmma32(load_a_frag(kiT + (size_t)(mt * 16) * CC + kcc, CC),
                   load_b_fragT(viT + (size_t)(et * 16) * CC + kcc, CC), acc);
#pragma unroll
    for (int j = 0; j < 8; ++j)
      dS[base + (size_t)(et * 16 + cn) * DD + mt * 16 + rbv + j] = acc[j];
  }
  if (tid < DD) {
    float s = 0.f;
    const _Float16* kr = kiT + tid * CC;
#pragma unroll 8
    for (int c2 = 0; c2 < CC; ++c2) s += (float)kr[c2];
    dz[((size_t)n * NBH + bh) * DD + tid] = s;
  }
}

// ---- kernel 2b: exclusive prefix of chunk states (tiny) ---------------------
// grid = NBH, block = 256
__global__ void la_prefix(const float* __restrict__ dS, const float* __restrict__ dz,
                          _Float16* __restrict__ Shx, float* __restrict__ zex) {
  const int bh  = blockIdx.x;
  const int tid = threadIdx.x;
  float acc[16];
#pragma unroll
  for (int j = 0; j < 16; ++j) acc[j] = 0.f;
  for (int n = 0; n < NCH; ++n) {
    size_t base = ((size_t)n * NBH + bh) * (DD * DD);
#pragma unroll
    for (int j = 0; j < 16; ++j) {
      int i = tid + j * 256;
      Shx[base + i] = (_Float16)acc[j];   // exclusive prefix, f16 for B frags
      acc[j] += dS[base + i];
    }
  }
  if (tid < DD) {
    float za = 0.f;
    for (int n = 0; n < NCH; ++n) {
      size_t zb = ((size_t)n * NBH + bh) * DD + tid;
      zex[zb] = za;
      za += dz[zb];
    }
  }
}

// ---- kernel 2c: per-chunk output (fully parallel) ---------------------------
// out = (q S_excl + masked(q k^T) v) / (q.z_excl + rowsum + eps)
// grid = (4 rowblocks, NCH, NBH), block = 256, dynamic LDS
//   STh f16 [64][64]  @0      (S^T exclusive, [e][d])
//   viT f16 [64][256] @8192
//   sbk f16 [64][256] @40960
//   den f32 [64]      @73728
//   zl  f32 [64]      @73984
#define AO_LDS_BYTES 74240
__global__ void la_attn_out(const _Float16* __restrict__ q,
                            const _Float16* __restrict__ k,
                            const _Float16* __restrict__ v,
                            const _Float16* __restrict__ Shx,
                            const float* __restrict__ zex,
                            _Float16* __restrict__ o) {
  extern __shared__ char smem[];
  _Float16* STh = (_Float16*)smem;
  _Float16* viT = (_Float16*)(smem + 8192);
  _Float16* sbk = (_Float16*)(smem + 40960);
  float*    den = (float*)(smem + 73728);
  float*    zl  = (float*)(smem + 73984);

  const int rb = blockIdx.x;
  const int n  = blockIdx.y;
  const int bh = blockIdx.z;
  const int b  = bh >> 4, h = bh & 15;
  const size_t cb = (size_t)b * LL * EE + (size_t)h * DD + (size_t)n * CC * EE;
  const _Float16* qc = q + cb;
  const _Float16* kc = k + cb;
  const _Float16* vc = v + cb;
  _Float16*       oh = o + cb;

  const int tid  = threadIdx.x;
  const int lane = tid & 31;
  const int wid  = tid >> 5;
  const int cn   = lane & 15;
  const int rbv  = (lane >> 4) * 8;
  const int r0   = rb * 64;

  {  // stage: S^T via async b128 copies; v transposed; z
    const _Float16* shp = Shx + ((size_t)n * NBH + bh) * (DD * DD);
#pragma unroll
    for (int i = tid; i < 512; i += 256)
      async_g2l_b128((uint32_t)(uintptr_t)STh + (uint32_t)i * 16,
                     (uint64_t)(uintptr_t)shp + (uint64_t)i * 16);
    const _Float16* vr = vc + (size_t)tid * EE;
#pragma unroll 8
    for (int d = 0; d < DD; ++d) viT[d * CC + tid] = vr[d];
    if (tid < DD) zl[tid] = zex[((size_t)n * NBH + bh) * DD + tid];
  }
  async_wait0();
  __syncthreads();

  // den = q . z_excl (per row of this block)
  if (tid < 64) {
    float accd = 0.f;
    const _Float16* qr = qc + (size_t)(r0 + tid) * EE;
#pragma unroll 8
    for (int d = 0; d < DD; ++d) accd += (float)qr[d] * zl[d];
    den[tid] = accd;
  }

  // scores = masked(q k^T), 4x16 tiles, 8 per wave
  for (int t = wid; t < 64; t += 8) {
    int mt = t >> 4, nt = t & 15;
    const _Float16* ab = qc + (size_t)(r0 + mt * 16) * EE;
    v8f acc = {};
    acc = wmma32(load_a_frag(ab, EE),
                 load_b_fragT(kc + (size_t)(nt * 16) * EE, EE), acc);
    acc = wmma32(load_a_frag(ab + 32, EE),
                 load_b_fragT(kc + (size_t)(nt * 16) * EE + 32, EE), acc);
    int rloc = mt * 16 + rbv;          // local row in block
    int cloc = nt * 16 + cn;           // column in chunk
#pragma unroll
    for (int j = 0; j < 8; ++j) {
      float sv = (cloc <= r0 + rloc + j) ? acc[j] : 0.f;
      sbk[(rloc + j) * CC + cloc] = (_Float16)sv;
    }
  }
  __syncthreads();

  if (tid < 64) {
    float rs = 0.f;
    const _Float16* sr = sbk + tid * CC;
#pragma unroll 8
    for (int c2 = 0; c2 < CC; ++c2) rs += (float)sr[c2];
    den[tid] += rs;
  }
  __syncthreads();

  // num = q S_excl + scores v ; out = num / (den + eps)
  for (int t = wid; t < 16; t += 8) {
    int mt = t >> 2, et = t & 3;
    const _Float16* ab = qc + (size_t)(r0 + mt * 16) * EE;
    v8f acc = {};
    acc = wmma32(load_a_frag(ab, EE),      load_b_fragT(STh + et * 16 * DD, DD), acc);
    acc = wmma32(load_a_frag(ab + 32, EE), load_b_fragT(STh + et * 16 * DD + 32, DD), acc);
#pragma unroll
    for (int ks = 0; ks < CC; ks += 32)
      acc = wmma32(load_a_frag(sbk + (mt * 16) * CC + ks, CC),
                   load_b_fragT(viT + (size_t)(et * 16) * CC + ks, CC), acc);
#pragma unroll
    for (int j = 0; j < 8; ++j) {
      int rl = mt * 16 + rbv + j;
      float ov = acc[j] / (den[rl] + EPSA);
      oh[(size_t)(r0 + rl) * EE + et * 16 + cn] = (_Float16)ov;
    }
  }
}

// ---- kernel 3: output projection + bias + residual --------------------------
// grid = (MM/128, EE/64), block = 256
__global__ void la_out_proj(const _Float16* __restrict__ a,
                            const float* __restrict__ W,
                            const float* __restrict__ bias,
                            const float* __restrict__ resid,
                            float* __restrict__ y) {
  __shared__ _Float16 At[128 * 32];
  __shared__ _Float16 Bt[64 * 32];
  const int row0 = blockIdx.x * 128;
  const int col0 = blockIdx.y * 64;
  const int tid  = threadIdx.x;
  const int lane = tid & 31;
  const int wid  = tid >> 5;
  const int mt2  = wid & 3;
  const int nh   = wid >> 2;

  v8f cacc[2][2] = {};
  for (int kb = 0; kb < EE; kb += 32) {
    {  // A already f16: async global->LDS, 32B per thread
      int r = tid >> 1, cg = (tid & 1) * 16;
      uint32_t l = (uint32_t)(uintptr_t)(At + r * 32 + cg);
      uint64_t g = (uint64_t)(uintptr_t)(a + (size_t)(row0 + r) * EE + kb + cg);
      async_g2l_b128(l, g);
      async_g2l_b128(l + 16, g + 16);
    }
    {
      int n = tid & 63, k0 = (tid >> 6) * 8;
#pragma unroll
      for (int i = 0; i < 8; ++i)
        Bt[n * 32 + k0 + i] = (_Float16)W[(size_t)(kb + k0 + i) * EE + col0 + n];
    }
    async_wait0();
    __syncthreads();
    v16h a0 = load_a_frag(At + (mt2 * 32) * 32, 32);
    v16h a1 = load_a_frag(At + (mt2 * 32 + 16) * 32, 32);
    v16h b0 = load_b_fragT(Bt + (nh * 32) * 32, 32);
    v16h b1 = load_b_fragT(Bt + (nh * 32 + 16) * 32, 32);
    cacc[0][0] = wmma32(a0, b0, cacc[0][0]);
    cacc[0][1] = wmma32(a0, b1, cacc[0][1]);
    cacc[1][0] = wmma32(a1, b0, cacc[1][0]);
    cacc[1][1] = wmma32(a1, b1, cacc[1][1]);
    __syncthreads();
  }

  const int cn  = lane & 15;
  const int rb2 = (lane >> 4) * 8;
#pragma unroll
  for (int mi = 0; mi < 2; ++mi) {
    int gr0 = row0 + (mt2 * 2 + mi) * 16 + rb2;
#pragma unroll
    for (int half = 0; half < 2; ++half) {
      const v8f cc = cacc[mi][half];
      int   gc = col0 + nh * 32 + half * 16 + cn;
      float bb = bias[gc];
#pragma unroll
      for (int j = 0; j < 8; ++j) {
        size_t idx = (size_t)(gr0 + j) * EE + gc;
        y[idx] = cc[j] + bb + resid[idx];
      }
    }
  }
}

// ---- kernel 4: LayerNorm per row -------------------------------------------
__global__ void la_layernorm(const float* __restrict__ y,
                             const float* __restrict__ gamma,
                             const float* __restrict__ beta,
                             float* __restrict__ out) {
  __shared__ float red[256];
  const int row = blockIdx.x;
  const int tid = threadIdx.x;
  const float* yr = y + (size_t)row * EE;

  float s = 0.f;
  for (int i = tid; i < EE; i += 256) s += yr[i];
  red[tid] = s; __syncthreads();
  for (int off = 128; off > 0; off >>= 1) {
    if (tid < off) red[tid] += red[tid + off];
    __syncthreads();
  }
  float mu = red[0] / (float)EE;
  __syncthreads();

  float vs = 0.f;
  for (int i = tid; i < EE; i += 256) { float d = yr[i] - mu; vs += d * d; }
  red[tid] = vs; __syncthreads();
  for (int off = 128; off > 0; off >>= 1) {
    if (tid < off) red[tid] += red[tid + off];
    __syncthreads();
  }
  float rstd = rsqrtf(red[0] / (float)EE + LNE);

  for (int i = tid; i < EE; i += 256)
    out[(size_t)row * EE + i] = gamma[i] * (yr[i] - mu) * rstd + beta[i];
}

// ---- launcher ---------------------------------------------------------------
extern "C" void kernel_launch(void* const* d_in, const int* in_sizes, int n_in,
                              void* d_out, int out_size, void* d_ws, size_t ws_size,
                              hipStream_t stream) {
  (void)in_sizes; (void)n_in; (void)out_size; (void)ws_size;
  const float* x     = (const float*)d_in[0];
  const float* Wq    = (const float*)d_in[1];
  const float* bq    = (const float*)d_in[2];
  const float* Wk    = (const float*)d_in[3];
  const float* bk    = (const float*)d_in[4];
  const float* Wv    = (const float*)d_in[5];
  const float* bv    = (const float*)d_in[6];
  const float* Wo    = (const float*)d_in[7];
  const float* bo    = (const float*)d_in[8];
  const float* gamma = (const float*)d_in[9];
  const float* beta  = (const float*)d_in[10];

  char* ws = (char*)d_ws;
  const size_t halfMat = (size_t)MM * EE * sizeof(_Float16);   // 32 MB
  const size_t stateF  = (size_t)NCH * NBH * DD * DD;          // 4 M elems
  const size_t zElems  = (size_t)NCH * NBH * DD;               // 64 K elems

  _Float16* qh  = (_Float16*)(ws);
  _Float16* kh  = (_Float16*)(ws + halfMat);
  _Float16* vh  = (_Float16*)(ws + 2 * halfMat);
  _Float16* ah  = (_Float16*)(ws + 3 * halfMat);
  float*    dS  = (float*)(ws + 4 * halfMat);
  float*    dz  = (float*)(ws + 4 * halfMat + stateF * 4);
  _Float16* Shx = (_Float16*)(ws + 4 * halfMat + stateF * 4 + zElems * 4);
  float*    zex = (float*)(ws + 4 * halfMat + stateF * 4 + zElems * 4 + stateF * 2);
  float*    yb  = (float*)(ws);   // reuses q/k region after attention

  dim3 gProj(MM / 128, EE / 64, 3);
  la_proj_qkv<<<gProj, 256, 0, stream>>>(x, Wq, bq, Wk, bk, Wv, bv, qh, kh, vh);

  dim3 gKV(NCH, NBH);
  la_chunk_kv<<<gKV, 256, KV_LDS_BYTES, stream>>>(kh, vh, dS, dz);

  la_prefix<<<NBH, 256, 0, stream>>>(dS, dz, Shx, zex);

  dim3 gAO(4, NCH, NBH);
  la_attn_out<<<gAO, 256, AO_LDS_BYTES, stream>>>(qh, kh, vh, Shx, zex, ah);

  dim3 gOut(MM / 128, EE / 64);
  la_out_proj<<<gOut, 256, 0, stream>>>(ah, Wo, bo, x, yb);

  la_layernorm<<<MM, 256, 0, stream>>>(yb, gamma, beta, (float*)d_out);
}
